// XcitAttention_9844065042498
// MI455X (gfx1250) — matrix operand order repaired and checked by reading, not compiled, not run on hardware
//
#include <hip/hip_runtime.h>
#include <hip/hip_bf16.h>

typedef __attribute__((ext_vector_type(16))) __bf16 v16bf;
typedef __attribute__((ext_vector_type(8)))  __bf16 v8bf;
typedef __attribute__((ext_vector_type(4)))  __bf16 v4bf;
typedef __attribute__((ext_vector_type(8)))  float  v8f;

// ---------------------------------------------------------------- helpers

__device__ __forceinline__ v8f wmma_bf16(v16bf a, v16bf b, v8f c) {
  // D = A(16x32 bf16) * B(32x16 bf16) + C(16x16 f32)
  return __builtin_amdgcn_wmma_f32_16x16x32_bf16(
      /*neg_a=*/false, a, /*neg_b=*/false, b,
      /*c_mod=*/(short)0, c, /*reuse_a=*/false, /*reuse_b=*/false);
}

__device__ __forceinline__ v8f vzero8() {
  v8f z;
#pragma unroll
  for (int i = 0; i < 8; ++i) z[i] = 0.0f;
  return z;
}

// A fragment: 16x32 bf16, row-major source [m][k] with leading dim lda.
// Layout used here: lanes 0-15 hold K = 0..15, lanes 16-31 hold K = 16..31 (contiguous).
__device__ __forceinline__ v16bf load_a_frag(const __bf16* A, int lda, int m0,
                                             int k0, int lane) {
  const int m  = m0 + (lane & 15);
  const int kb = k0 + ((lane >> 4) << 4);  // +0 or +16
  const __bf16* p = A + (size_t)m * lda + kb;
  v8bf lo = *(const v8bf*)(p);        // K = kb .. kb+7
  v8bf hi = *(const v8bf*)(p + 8);    // K = kb+8 .. kb+15
  v16bf a;
#pragma unroll
  for (int i = 0; i < 8; ++i) { a[i] = lo[i]; a[i + 8] = hi[i]; }
  return a;
}

// B fragment: 32x16 bf16, source stored n-major [n][k] with leading dim ldb.
// Layout used here: lanes 0-15 hold K = 0..15, lanes 16-31 hold K = 16..31 (contiguous).
__device__ __forceinline__ v16bf load_b_frag(const __bf16* Bp, int ldb, int n0,
                                             int k0, int lane) {
  const int n  = n0 + (lane & 15);
  const int kb = k0 + ((lane >> 4) << 4);  // +0 or +16
  return *(const v16bf*)(Bp + (size_t)n * ldb + kb);
}

// ---------------------------------------------------------------- convert
// Vectorized f32 -> bf16: 16B load, 8B store per thread-iteration.

__global__ void cvt_f32_bf16_kernel(const float4* __restrict__ src,
                                    v4bf* __restrict__ dst, size_t n4) {
  size_t i = (size_t)blockIdx.x * blockDim.x + threadIdx.x;
  size_t stride = (size_t)gridDim.x * blockDim.x;
  for (; i < n4; i += stride) {
    float4 f = src[i];
    v4bf o;
    o[0] = (__bf16)f.x; o[1] = (__bf16)f.y; o[2] = (__bf16)f.z; o[3] = (__bf16)f.w;
    dst[i] = o;
  }
}

// ---------------------------------------------------------------- qkv gemm
// C[M=32768, N=3072] = X[M,1024] * Wqkv^T ; epilogue scatters (transposed)
// into qkvT[s][b][h][d][n] (bf16), n (token) contiguous.

__global__ __launch_bounds__(256) void gemm_qkv_kernel(
    const __bf16* __restrict__ X, const __bf16* __restrict__ W,
    __bf16* __restrict__ qkvT) {
  __shared__ __bf16 tile[128 * 136];  // [col][token], padded stride
  const int tid  = threadIdx.x;
  const int lane = tid & 31;
  const int w    = tid >> 5;
  const int wm   = w >> 1;   // 0..3  (32 rows each)
  const int wn   = w & 1;    // 0..1  (64 cols each)
  const int row0 = blockIdx.y * 128;  // token-row block (within one b)
  const int col0 = blockIdx.x * 128;  // output column block

  v8f acc[2][4];
#pragma unroll
  for (int i = 0; i < 2; ++i)
#pragma unroll
    for (int j = 0; j < 4; ++j) acc[i][j] = vzero8();

  for (int k0 = 0; k0 < 1024; k0 += 32) {
    // prefetch next-next K tile (speculative; silently dropped if OOB)
    {
      const int kp = k0 + 64;
      const int m  = row0 + wm * 32 + (lane & 15);
      const int nn = col0 + wn * 64 + (lane & 15);
      __builtin_prefetch(X + (size_t)m * 1024 + kp, 0, 1);
      __builtin_prefetch(W + (size_t)nn * 1024 + kp, 0, 1);
    }
    v16bf a0 = load_a_frag(X, 1024, row0 + wm * 32, k0, lane);
    v16bf a1 = load_a_frag(X, 1024, row0 + wm * 32 + 16, k0, lane);
#pragma unroll
    for (int j = 0; j < 4; ++j) {
      v16bf bf = load_b_frag(W, 1024, col0 + wn * 64 + j * 16, k0, lane);
      acc[0][j] = wmma_bf16(a0, bf, acc[0][j]);
      acc[1][j] = wmma_bf16(a1, bf, acc[1][j]);
    }
  }

  // stage transposed into LDS: tile[col][token]
  const int ln = lane & 15;
  const int hi = (lane >> 4) * 8;
#pragma unroll
  for (int i = 0; i < 2; ++i)
#pragma unroll
    for (int j = 0; j < 4; ++j)
#pragma unroll
      for (int r = 0; r < 8; ++r) {
        int tok = wm * 32 + i * 16 + hi + r;
        int c   = wn * 64 + j * 16 + ln;
        tile[c * 136 + tok] = (__bf16)acc[i][j][r];
      }
  __syncthreads();

  // each thread stores 64 contiguous tokens of one output column
  const int c    = tid & 127;
  const int hh   = tid >> 7;
  const int col  = col0 + c;
  const int s    = col >> 10;
  const int rem  = col & 1023;
  const int h    = rem >> 6;
  const int d    = rem & 63;
  const int b_   = row0 >> 12;    // row0 / 4096
  const int tok0 = row0 & 4095;
  const size_t dstrow = ((size_t)((s * 8 + b_) * 16 + h)) * 64 + d;
  const uint4* src = (const uint4*)&tile[c * 136 + hh * 64];
  uint4* dst = (uint4*)(qkvT + dstrow * 4096 + tok0 + hh * 64);
#pragma unroll
  for (int i = 0; i < 8; ++i) dst[i] = src[i];
}

// ---------------------------------------------------------------- attention
// One block per (b,h). S = q k^T (K=4096 WMMA), scaled by TEMP/(|q||k|),
// masked softmax, out = P @ V with V transposed through LDS;
// epilogue writes attout[B,N,C] bf16 (token-major).

__global__ __launch_bounds__(256) void attn_kernel(
    const __bf16* __restrict__ qkvT, const unsigned char* __restrict__ mask,
    __bf16* __restrict__ attout) {
  __shared__ float  Sl[64 * 68];
  __shared__ __bf16 Pl[64 * 72];
  __shared__ __bf16 VL[256 * 80];   // also reused for output staging
  __shared__ float  partial[256];
  __shared__ float  rnorm[128];     // [0..63] = 1/max(|q_d|,eps), [64..127] = k

  const int bh = blockIdx.x;
  const int b  = bh >> 4;
  const int h  = bh & 15;
  const __bf16* q = qkvT + (size_t)(bh)*64 * 4096;
  const __bf16* k = qkvT + (size_t)(128 + bh) * 64 * 4096;
  const __bf16* v = qkvT + (size_t)(256 + bh) * 64 * 4096;
  const int tid  = threadIdx.x;
  const int lane = tid & 31;
  const int w    = tid >> 5;

  // ---- row L2 norms of q and k (over 4096 tokens)
  {
    const int r = tid >> 1;
    const int half = tid & 1;
    const __bf16* rp =
        (r < 64 ? q + (size_t)r * 4096 : k + (size_t)(r - 64) * 4096) +
        half * 2048;
    float s = 0.0f;
    for (int i = 0; i < 2048; i += 8) {
      v8bf xv = *(const v8bf*)(rp + i);
#pragma unroll
      for (int j = 0; j < 8; ++j) { float f = (float)xv[j]; s += f * f; }
    }
    partial[tid] = s;
  }
  __syncthreads();
  if (tid < 128) {
    float ss = partial[2 * tid] + partial[2 * tid + 1];
    rnorm[tid] = 1.0f / fmaxf(sqrtf(ss), 1e-12f);
  }
  __syncthreads();

  // ---- S = q k^T  (64x64, K = 4096)
  {
    const int mi = w >> 1;          // 0..3
    const int nb = (w & 1) * 32;    // 0 or 32
    v8f s0 = vzero8(), s1 = vzero8();
    for (int k0 = 0; k0 < 4096; k0 += 32) {
      v16bf a  = load_a_frag(q, 4096, mi * 16, k0, lane);
      v16bf b0 = load_b_frag(k, 4096, nb, k0, lane);
      v16bf b1 = load_b_frag(k, 4096, nb + 16, k0, lane);
      s0 = wmma_bf16(a, b0, s0);
      s1 = wmma_bf16(a, b1, s1);
    }
    const int ln = lane & 15;
    const int hi = (lane >> 4) * 8;
#pragma unroll
    for (int r = 0; r < 8; ++r) {
      int m = mi * 16 + hi + r;
      Sl[m * 68 + nb + ln]      = s0[r];
      Sl[m * 68 + nb + 16 + ln] = s1[r];
    }
  }
  __syncthreads();

  // ---- scale, mask, softmax -> Pl (bf16)
  if (tid < 64) {
    const int d = tid;
    const float rq = 0.8f * rnorm[d];  // TEMP folded in
    const unsigned char* mp = mask + ((size_t)bh * 64 + d) * 64;
    float mx = -1e30f;
    for (int e = 0; e < 64; ++e) {
      float val = Sl[d * 68 + e] * (rq * rnorm[64 + e]);
      val = mp[e] ? val : -1.0e6f;
      Sl[d * 68 + e] = val;
      mx = fmaxf(mx, val);
    }
    float sum = 0.0f;
    for (int e = 0; e < 64; ++e) {
      float p = __expf(Sl[d * 68 + e] - mx);
      Sl[d * 68 + e] = p;
      sum += p;
    }
    float inv = 1.0f / sum;
    for (int e = 0; e < 64; ++e) Pl[d * 72 + e] = (__bf16)(Sl[d * 68 + e] * inv);
  }
  __syncthreads();

  // ---- out = P(64x64) @ V(64x4096), 256-token chunks
  for (int c0 = 0; c0 < 4096; c0 += 256) {
    // stage V chunk transposed: VL[token][e]
    {
      const int e  = tid & 63;
      const int tb = (tid >> 6) * 64;
#pragma unroll
      for (int jj = 0; jj < 64; jj += 8) {
        v8bf xv = *(const v8bf*)(v + (size_t)e * 4096 + c0 + tb + jj);
#pragma unroll
        for (int u = 0; u < 8; ++u) VL[(tb + jj + u) * 80 + e] = xv[u];
      }
    }
    __syncthreads();

    v8f acc[2][4];
#pragma unroll
    for (int sb = 0; sb < 2; ++sb)
#pragma unroll
      for (int mi = 0; mi < 4; ++mi) acc[sb][mi] = vzero8();

#pragma unroll
    for (int kk = 0; kk < 64; kk += 32) {
      v16bf a[4];
#pragma unroll
      for (int mi = 0; mi < 4; ++mi) a[mi] = load_a_frag(Pl, 72, mi * 16, kk, lane);
#pragma unroll
      for (int sb = 0; sb < 2; ++sb) {
        const int tb = (w * 2 + sb) * 16;
        v16bf bf = load_b_frag(VL, 80, tb, kk, lane);
#pragma unroll
        for (int mi = 0; mi < 4; ++mi) acc[sb][mi] = wmma_bf16(a[mi], bf, acc[sb][mi]);
      }
    }
    __syncthreads();  // all VL reads done before overwrite

    // stage output transposed: VL[token][d]
    {
      const int ln = lane & 15;
      const int hi = (lane >> 4) * 8;
#pragma unroll
      for (int sb = 0; sb < 2; ++sb)
#pragma unroll
        for (int mi = 0; mi < 4; ++mi)
#pragma unroll
          for (int r = 0; r < 8; ++r) {
            int tok = (w * 2 + sb) * 16 + ln;
            int dd  = mi * 16 + hi + r;
            VL[tok * 80 + dd] = (__bf16)acc[sb][mi][r];
          }
    }
    __syncthreads();

    // each thread writes one token's 64 channels (128 B contiguous)
    {
      const int n = c0 + tid;
      uint4* dst = (uint4*)(attout + ((size_t)(b * 4096 + n)) * 1024 + h * 64);
      const uint4* src = (const uint4*)&VL[tid * 80];
#pragma unroll
      for (int i = 0; i < 8; ++i) dst[i] = src[i];
    }
    __syncthreads();
  }
}

// ---------------------------------------------------------------- proj gemm
// out[M=32768,1024] = A[M,1024] * Wproj^T + bias  (f32 output)
// Epilogue stages the 128x128 f32 tile in LDS and emits coalesced b128
// stores (d_out is 134 MB; only the qkvT write is larger).

__global__ __launch_bounds__(256) void gemm_proj_kernel(
    const __bf16* __restrict__ A, const __bf16* __restrict__ W,
    const float* __restrict__ bias, float* __restrict__ out) {
  __shared__ float tile[128 * 132];  // [row][col], padded stride
  const int tid  = threadIdx.x;
  const int lane = tid & 31;
  const int w    = tid >> 5;
  const int wm   = w >> 1;
  const int wn   = w & 1;
  const int row0 = blockIdx.y * 128;
  const int col0 = blockIdx.x * 128;

  v8f acc[2][4];
#pragma unroll
  for (int i = 0; i < 2; ++i)
#pragma unroll
    for (int j = 0; j < 4; ++j) acc[i][j] = vzero8();

  for (int k0 = 0; k0 < 1024; k0 += 32) {
    {
      const int kp = k0 + 64;
      const int m  = row0 + wm * 32 + (lane & 15);
      const int nn = col0 + wn * 64 + (lane & 15);
      __builtin_prefetch(A + (size_t)m * 1024 + kp, 0, 1);
      __builtin_prefetch(W + (size_t)nn * 1024 + kp, 0, 1);
    }
    v16bf a0 = load_a_frag(A, 1024, row0 + wm * 32, k0, lane);
    v16bf a1 = load_a_frag(A, 1024, row0 + wm * 32 + 16, k0, lane);
#pragma unroll
    for (int j = 0; j < 4; ++j) {
      v16bf bf = load_b_frag(W, 1024, col0 + wn * 64 + j * 16, k0, lane);
      acc[0][j] = wmma_bf16(a0, bf, acc[0][j]);
      acc[1][j] = wmma_bf16(a1, bf, acc[1][j]);
    }
  }

  // stage into LDS: tile[row][col]
  const int ln = lane & 15;
  const int hi = (lane >> 4) * 8;
#pragma unroll
  for (int i = 0; i < 2; ++i)
#pragma unroll
    for (int j = 0; j < 4; ++j)
#pragma unroll
      for (int r = 0; r < 8; ++r) {
        const int row = wm * 32 + i * 16 + hi + r;
        const int col = wn * 64 + j * 16 + ln;
        tile[row * 132 + col] = acc[i][j][r];
      }
  __syncthreads();

  // each thread writes 64 contiguous f32 of one output row (+bias)
  const int r  = tid >> 1;
  const int hh = tid & 1;
  const int row = row0 + r;
  const float4* srcp = (const float4*)&tile[r * 132 + hh * 64];
  const float4* bp   = (const float4*)&bias[col0 + hh * 64];
  float4* dstp = (float4*)&out[(size_t)row * 1024 + col0 + hh * 64];
#pragma unroll
  for (int i = 0; i < 16; ++i) {
    float4 vv = srcp[i];
    float4 bb = bp[i];
    vv.x += bb.x; vv.y += bb.y; vv.z += bb.z; vv.w += bb.w;
    dstp[i] = vv;
  }
}

// ---------------------------------------------------------------- launch

extern "C" void kernel_launch(void* const* d_in, const int* in_sizes, int n_in,
                              void* d_out, int out_size, void* d_ws,
                              size_t ws_size, hipStream_t stream) {
  (void)in_sizes; (void)n_in; (void)out_size; (void)ws_size;
  const float* x     = (const float*)d_in[0];
  const unsigned char* mask = (const unsigned char*)d_in[1];
  const float* wqkv  = (const float*)d_in[2];
  const float* wproj = (const float*)d_in[3];
  const float* bproj = (const float*)d_in[4];
  float* out = (float*)d_out;

  char* ws = (char*)d_ws;
  const size_t XB_BYTES    = (size_t)32768 * 1024 * 2;  // 64 MB
  const size_t WQKV_BYTES  = (size_t)3072 * 1024 * 2;   // 6 MB
  const size_t WPROJ_BYTES = (size_t)1024 * 1024 * 2;   // 2 MB
  const size_t QKVT_BYTES  = (size_t)3 * 8 * 16 * 64 * 4096 * 2;  // 192 MB
  __bf16* xb     = (__bf16*)(ws);
  __bf16* wqb    = (__bf16*)(ws + XB_BYTES);
  __bf16* wpb    = (__bf16*)(ws + XB_BYTES + WQKV_BYTES);
  __bf16* qkvT   = (__bf16*)(ws + XB_BYTES + WQKV_BYTES + WPROJ_BYTES);
  __bf16* attout = (__bf16*)(ws + XB_BYTES + WQKV_BYTES + WPROJ_BYTES + QKVT_BYTES);

  cvt_f32_bf16_kernel<<<2048, 256, 0, stream>>>((const float4*)x, (v4bf*)xb,
                                                (size_t)32768 * 1024 / 4);
  cvt_f32_bf16_kernel<<<512, 256, 0, stream>>>((const float4*)wqkv, (v4bf*)wqb,
                                               (size_t)3072 * 1024 / 4);
  cvt_f32_bf16_kernel<<<256, 256, 0, stream>>>((const float4*)wproj, (v4bf*)wpb,
                                               (size_t)1024 * 1024 / 4);

  gemm_qkv_kernel<<<dim3(24, 256), 256, 0, stream>>>(xb, wqb, qkvT);
  attn_kernel<<<dim3(128), 256, 0, stream>>>(qkvT, mask, attout);
  gemm_proj_kernel<<<dim3(8, 256), 256, 0, stream>>>(attout, wpb, bproj, out);
}
